// CausalSelfAttentionWithCache_54348516163982
// MI455X (gfx1250) — matrix operand order repaired and checked by reading, not compiled, not run on hardware
//
#include <hip/hip_runtime.h>

// CausalSelfAttentionWithCache for MI455X (gfx1250, wave32, WMMA bf16).
// GEMM tiles staged by the Tensor Data Mover (TENSOR_LOAD_TO_LDS, TENSORcnt,
// hardware row-padding -> bank-conflict-free pitch); attention K chunks staged
// by GLOBAL_LOAD_ASYNC_TO_LDS_B128 (ASYNCcnt). Both double-buffered.
// Workspace layout (bytes), total ~151MB:
//   xb    @ 0          : bf16(x)            16,777,216
//   wqkv  @ 16777216   : bf16(qkv_w)        25,165,824
//   wout  @ 41943040   : bf16(out_w)         8,388,608
//   qb    @ 50331648   : bf16 q [B,H,S,HD]  16,777,216   (pre-scaled by 1/sqrt(HD))
//   kb    @ 67108864   : bf16 k [B,H,Skv,HD]33,554,432
//   vb    @ 100663296  : bf16 v [B,H,Skv,HD]33,554,432
//   attnb @ 134217728  : bf16 attn [B,S,D]  16,777,216

typedef __attribute__((ext_vector_type(16))) __bf16 v16bf;
typedef __attribute__((ext_vector_type(8)))  __bf16 v8bf;
typedef __attribute__((ext_vector_type(4)))  __bf16 v4bf;
typedef __attribute__((ext_vector_type(8)))  float  v8f;
typedef __attribute__((ext_vector_type(4)))  unsigned u32x4;
typedef __attribute__((ext_vector_type(8)))  int      i32x8;
typedef __attribute__((ext_vector_type(4)))  int      i32x4;

#if defined(__gfx1250__) && __has_builtin(__builtin_amdgcn_tensor_load_to_lds)
#define GEMM_TDM 1
#else
#define GEMM_TDM 0
#endif

__device__ __forceinline__ __bf16 f2bf(float f) {
  union { float f; unsigned u; } x; x.f = f;
  unsigned r = x.u + 0x7FFFu + ((x.u >> 16) & 1u);   // round-to-nearest-even
  unsigned short hsh = (unsigned short)(r >> 16);
  __bf16 o;
  __builtin_memcpy(&o, &hsh, 2);
  return o;
}

// Low 32 bits of the flat address of a __shared__ object are the wave-relative
// LDS byte offset (ISA 10.2: LDS aperture truncates to addr[31:0]).
__device__ __forceinline__ unsigned lds_off(const void* p) {
  return (unsigned)(size_t)p;
}

// 16B per lane, global -> LDS, tracked by ASYNCcnt (no data VGPRs touched).
__device__ __forceinline__ void async_load_b128(unsigned ldsoff, const void* gaddr) {
  asm volatile("global_load_async_to_lds_b128 %0, %1, off"
               :: "v"(ldsoff), "v"(gaddr) : "memory");
}

#if GEMM_TDM
// TDM 2D tile load: tile_w=32 bf16 per row (64B = 16 DWORDs), tile_h rows,
// row stride `stride_elems` (bf16 units). pad_enable inserts 4 DWORDs (16B =
// 8 bf16) after every 16 DWORDs stored -> LDS pitch 40 elements (80B), i.e.
// the bank-conflict-free layout, synthesized by the DMA engine itself.
// D# bitfields per ISA 8.3 (group0) / 8.4 (group1); groups 2-3 zero (2D).
__device__ __forceinline__ void tdm_load_tile_2d(const void* gaddr, unsigned ldsaddr,
                                                 unsigned tile_h, unsigned stride_elems) {
  const unsigned long long ga = (unsigned long long)(size_t)gaddr;
  u32x4 g0;
  g0[0] = 1u;                                   // count=1, user descriptor
  g0[1] = ldsaddr;                              // lds_addr
  g0[2] = (unsigned)ga;                         // global_addr[31:0]
  g0[3] = (unsigned)(ga >> 32) | (2u << 30);    // global_addr[56:32] | type=2
  i32x8 g1;
  g1[0] = (int)((1u << 16) |                    // data_size=1 (2 bytes)
                (1u << 20) |                    // pad_enable
                (3u << 22) |                    // pad_interval: 16 DWORDs
                (3u << 25));                    // pad_amount:   4 DWORDs
  g1[1] = (int)(32u << 16);                     // tensor_dim0 = 32 (bits 79:48)
  g1[2] = (int)((tile_h & 0xFFFFu) << 16);      // tensor_dim1 (bits 111:80)
  g1[3] = (int)(32u << 16);                     // tile_dim0 = 32 (bits 127:112)
  g1[4] = (int)(tile_h & 0xFFFFu);              // tile_dim1 | tile_dim2=0
  g1[5] = (int)stride_elems;                    // tensor_dim0_stride[31:0]
  g1[6] = 0;                                    // stride0 hi | stride1 lo
  g1[7] = 0;
  const i32x4 z = {};
#if __clang_major__ >= 23
  const i32x8 z8 = {};                          // therock/clang-23: 6-arg form
  __builtin_amdgcn_tensor_load_to_lds(g0, g1, z, z, z8, 0);
#else
  __builtin_amdgcn_tensor_load_to_lds(g0, g1, z, z, 0);   // ROCm 7.2 / clang-22
#endif
}
#endif

// A-fragment for V_WMMA_F32_16X16X32_BF16 (16x32, MxK).
// ISA layout: lanes 0-15 hold K=[h8..h8+7] and K=[16+h8..16+h8+7] with h8=8*(lane/16).
__device__ __forceinline__ v16bf load_a_frag(const __bf16* base, int ld, int lane) {
  const int m  = lane & 15;
  const int h8 = (lane >> 4) << 3;
  const __bf16* p = base + m * ld + h8;
  v8bf lo = *(const v8bf*)(p);
  v8bf hi = *(const v8bf*)(p + 16);
  v16bf r;
#pragma unroll
  for (int i = 0; i < 8; ++i) { r[i] = lo[i]; r[i + 8] = hi[i]; }
  return r;
}

// B-fragment (32x16, KxN) where memory holds B transposed: row n (stride ld)
// contains B[k][n], k contiguous. ISA: lanes hold K=[h8..h8+7] and K=[16+h8..16+h8+7], h8=8*(lane/16).
__device__ __forceinline__ v16bf load_bt_frag(const __bf16* base, int ld, int lane) {
  const int n   = lane & 15;
  const int k16 = (lane >> 4) << 3;
  const __bf16* p = base + n * ld + k16;
  v8bf lo = *(const v8bf*)(p);
  v8bf hi = *(const v8bf*)(p + 16);
  v16bf r;
#pragma unroll
  for (int i = 0; i < 8; ++i) { r[i] = lo[i]; r[i + 8] = hi[i]; }
  return r;
}

__global__ void f32_to_bf16_kernel(const float* __restrict__ src,
                                   __bf16* __restrict__ dst, int n4) {
  for (int i = blockIdx.x * blockDim.x + threadIdx.x; i < n4;
       i += gridDim.x * blockDim.x) {
    const float4 v = ((const float4*)src)[i];
    v4bf o; o[0] = f2bf(v.x); o[1] = f2bf(v.y); o[2] = f2bf(v.z); o[3] = f2bf(v.w);
    ((v4bf*)dst)[i] = o;
  }
}

// past_[kv] [B,H,2048,128] -> fp32 cache rows 0..2047 of [B,H,4096,128] + bf16 mirror
__global__ void copy_past_kernel(const float* __restrict__ src,
                                 float* __restrict__ dstf,
                                 __bf16* __restrict__ dstb, int n4) {
  for (int i = blockIdx.x * blockDim.x + threadIdx.x; i < n4;
       i += gridDim.x * blockDim.x) {
    const int bh = i >> 16;             // 2048*128/4 = 65536 float4 per (b,h)
    const int r4 = i & 65535;
    const float4 v = ((const float4*)src)[i];
    const int dof = bh * 131072 + r4;   // 4096*128/4 per (b,h) in dest
    ((float4*)dstf)[dof] = v;
    v4bf o; o[0] = f2bf(v.x); o[1] = f2bf(v.y); o[2] = f2bf(v.z); o[3] = f2bf(v.w);
    ((v4bf*)dstb)[dof] = o;
  }
}

// C[M,N] = A[M,K] * W[N,K]^T + bias.  Block tile 128x128 (8 waves, each 32x64).
// Double-buffered LDS tiles filled by the TDM (2 TENSOR_LOAD_TO_LDS per tile,
// issued by wave 0; s_wait_tensorcnt 2 = "previous tile landed"), so tile i+1's
// DMA overlaps tile i's WMMAs. Fallback: per-lane async-to-LDS loads.
// MODE 0: QKV projection -> scatter q/k/v.  MODE 1: output projection -> fp32 out.
template <int MODE>
__global__ __launch_bounds__(256) void gemm_bf16_kernel(
    const __bf16* __restrict__ A, const __bf16* __restrict__ W,
    const float* __restrict__ bias,
    float* __restrict__ outp, float* __restrict__ kout, float* __restrict__ vout,
    __bf16* __restrict__ qb, __bf16* __restrict__ kb, __bf16* __restrict__ vb,
    int K) {
  __shared__ __align__(16) __bf16 At[2][128 * 40];  // 128 x 32, pitch 40 (80B)
  __shared__ __align__(16) __bf16 Bt[2][128 * 40];
  const int tid  = threadIdx.x;
  const int wave = tid >> 5, lane = tid & 31;
  const int wm = wave >> 1, wn = wave & 1;
  const int gm = blockIdx.y * 128, gn = blockIdx.x * 128;
  const int iters = K >> 5;
  v8f acc[2][4] = {};
#if GEMM_TDM
  if (wave == 0) {
    tdm_load_tile_2d(A + (size_t)gm * K, lds_off(&At[0][0]), 128, K);
    tdm_load_tile_2d(W + (size_t)gn * K, lds_off(&Bt[0][0]), 128, K);
  }
#else
  const int lrow = tid >> 1;        // 0..127
  const int lk   = (tid & 1) << 4;  // 0 or 16
  const __bf16* pa0 = A + (size_t)(gm + lrow) * K + lk;
  const __bf16* pb0 = W + (size_t)(gn + lrow) * K + lk;
  const unsigned aoff = lrow * 40 + lk;
  async_load_b128(lds_off(&At[0][aoff]),     pa0);
  async_load_b128(lds_off(&At[0][aoff + 8]), pa0 + 8);
  async_load_b128(lds_off(&Bt[0][aoff]),     pb0);
  async_load_b128(lds_off(&Bt[0][aoff + 8]), pb0 + 8);
#endif
  for (int it = 0; it < iters; ++it) {
    const int cur = it & 1, nxt = cur ^ 1;
#if GEMM_TDM
    if (wave == 0) {
      if (it + 1 < iters) {
        const int k1 = (it + 1) << 5;
        tdm_load_tile_2d(A + (size_t)gm * K + k1, lds_off(&At[nxt][0]), 128, K);
        tdm_load_tile_2d(W + (size_t)gn * K + k1, lds_off(&Bt[nxt][0]), 128, K);
        __builtin_amdgcn_s_wait_tensorcnt(2);  // tile `it` (2 older TDM ops) landed
      } else {
        __builtin_amdgcn_s_wait_tensorcnt(0);
      }
    }
#else
    if (it + 1 < iters) {
      const int k1 = (it + 1) << 5;
      async_load_b128(lds_off(&At[nxt][aoff]),     pa0 + k1);
      async_load_b128(lds_off(&At[nxt][aoff + 8]), pa0 + k1 + 8);
      async_load_b128(lds_off(&Bt[nxt][aoff]),     pb0 + k1);
      async_load_b128(lds_off(&Bt[nxt][aoff + 8]), pb0 + k1 + 8);
      asm volatile("s_wait_asynccnt 0x4" ::: "memory");
    } else {
      asm volatile("s_wait_asynccnt 0x0" ::: "memory");
    }
#endif
    __syncthreads();  // tile `it` visible block-wide
    v16bf af[2], bfr[4];
#pragma unroll
    for (int mi = 0; mi < 2; ++mi)
      af[mi] = load_a_frag(&At[cur][(wm * 32 + mi * 16) * 40], 40, lane);
#pragma unroll
    for (int ni = 0; ni < 4; ++ni)
      bfr[ni] = load_bt_frag(&Bt[cur][(wn * 64 + ni * 16) * 40], 40, lane);
#pragma unroll
    for (int mi = 0; mi < 2; ++mi)
#pragma unroll
      for (int ni = 0; ni < 4; ++ni)
        acc[mi][ni] = __builtin_amdgcn_wmma_f32_16x16x32_bf16(
            false, af[mi], false, bfr[ni], (short)0, acc[mi][ni], false, false);
    __syncthreads();  // all reads of buf[cur] done; next iter may overwrite it
  }
  const int n = lane & 15, half = lane >> 4;
#pragma unroll
  for (int mi = 0; mi < 2; ++mi) {
#pragma unroll
    for (int ni = 0; ni < 4; ++ni) {
      const int r0  = gm + wm * 32 + mi * 16;
      const int col = gn + wn * 64 + ni * 16 + n;
      const float bv = bias[col];
      if (MODE == 0) {
        const int sec = col >> 11, d = col & 2047, h = d >> 7, hd = d & 127;
#pragma unroll
        for (int i = 0; i < 8; ++i) {
          const float v = acc[mi][ni][i] + bv;
          const int r = r0 + half * 8 + i;
          const int b = r >> 11, s = r & 2047;
          if (sec == 0) {
            // q: pre-scale by 1/sqrt(HD)=1/sqrt(128)
            qb[((b * 16 + h) * 2048 + s) * 128 + hd] = f2bf(v * 0.08838834764831845f);
          } else {
            const int idx = ((b * 16 + h) * 4096 + 2048 + s) * 128 + hd;
            if (sec == 1) { kout[idx] = v; kb[idx] = f2bf(v); }
            else          { vout[idx] = v; vb[idx] = f2bf(v); }
          }
        }
      } else {
#pragma unroll
        for (int i = 0; i < 8; ++i) {
          const float v = acc[mi][ni][i] + bv;
          const int r = r0 + half * 8 + i;
          outp[(size_t)r * 2048 + col] = v;
        }
      }
    }
  }
}

// Flash attention: block = 8 waves = 128 q rows of one (b,h); kv chunks of 32.
// K chunk double-buffered via async-to-LDS DMA (overlaps the whole compute
// phase of the previous chunk); V^T staged with a VALU transpose (scatter the
// async unit cannot do), overlapping the in-flight K DMA.
__global__ __launch_bounds__(256) void attn_kernel(
    const __bf16* __restrict__ qb, const __bf16* __restrict__ kb,
    const __bf16* __restrict__ vb, __bf16* __restrict__ attnb) {
  __shared__ __align__(16) __bf16 Kt[2][32 * 136];  // [kv 32][hd 128] pitch 136
  __shared__ __align__(16) __bf16 VtT[128 * 40];    // [hd 128][kv 32] pitch 40
  __shared__ __align__(16) __bf16 Pt[8][16 * 40];   // per-wave P tile [16][32] pitch 40
  const int tid = threadIdx.x, wave = tid >> 5, lane = tid & 31;
  const int b = blockIdx.z, h = blockIdx.y, bh = b * 16 + h;
  const int q0 = blockIdx.x * 128 + wave * 16;
  const __bf16* qbh = qb + (size_t)bh * 2048 * 128 + (size_t)q0 * 128;
  const __bf16* kbh = kb + (size_t)bh * 4096 * 128;
  const __bf16* vbh = vb + (size_t)bh * 4096 * 128;
  v16bf qf[4];
#pragma unroll
  for (int kk = 0; kk < 4; ++kk) qf[kk] = load_a_frag(qbh + kk * 32, 128, lane);
  v8f oacc[8] = {};
  float mrow[8], lrow[8];
#pragma unroll
  for (int i = 0; i < 8; ++i) { mrow[i] = -1e30f; lrow[i] = 0.f; }
  const int cr = tid >> 3;        // cooperative-load kv row 0..31
  const int cc = (tid & 7) << 4;  // hd offset 0,16,..,112
  const int n = lane & 15, half = lane >> 4;
  const unsigned koff = cr * 136 + cc;
  // prologue: stage K chunk 0 into buffer 0
  {
    const __bf16* pk = kbh + (size_t)cr * 128 + cc;
    async_load_b128(lds_off(&Kt[0][koff]),     pk);
    async_load_b128(lds_off(&Kt[0][koff + 8]), pk + 8);
  }
  for (int it = 0; it < 128; ++it) {
    const int kv0 = it << 5;
    const int cur = it & 1, nxt = cur ^ 1;
    __syncthreads();  // prev chunk's reads of VtT/Kt[nxt] complete
    {
      const __bf16* pv = vbh + (size_t)(kv0 + cr) * 128 + cc;
      v8bf v0v = *(const v8bf*)pv;
      v8bf v1v = *(const v8bf*)(pv + 8);
#pragma unroll
      for (int j = 0; j < 8; ++j) {
        VtT[(cc + j) * 40 + cr]     = v0v[j];
        VtT[(cc + 8 + j) * 40 + cr] = v1v[j];
      }
    }
    if (it + 1 < 128) {
      const __bf16* pk = kbh + (size_t)(kv0 + 32 + cr) * 128 + cc;
      async_load_b128(lds_off(&Kt[nxt][koff]),     pk);
      async_load_b128(lds_off(&Kt[nxt][koff + 8]), pk + 8);
      asm volatile("s_wait_asynccnt 0x2" ::: "memory");  // K chunk `it` landed
    } else {
      asm volatile("s_wait_asynccnt 0x0" ::: "memory");
    }
    __syncthreads();  // K(it) + V(it) visible block-wide
    // S[16x32] = Q[16x128] * K_chunk^T, accumulated over 4 k-steps
    v8f s0 = {}, s1 = {};
#pragma unroll
    for (int kk = 0; kk < 4; ++kk) {
      v16bf bk0 = load_bt_frag(&Kt[cur][kk * 32], 136, lane);
      v16bf bk1 = load_bt_frag(&Kt[cur][16 * 136 + kk * 32], 136, lane);
      s0 = __builtin_amdgcn_wmma_f32_16x16x32_bf16(false, qf[kk], false, bk0, (short)0, s0, false, false);
      s1 = __builtin_amdgcn_wmma_f32_16x16x32_bf16(false, qf[kk], false, bk1, (short)0, s1, false, false);
    }
    // Online softmax: C-tile row m=8*half+i lives in the lane's 16-lane group.
    float scale[8], p0[8], p1[8];
#pragma unroll
    for (int i = 0; i < 8; ++i) {
      float mx = fmaxf(s0[i], s1[i]);
#pragma unroll
      for (int msk = 1; msk < 16; msk <<= 1) mx = fmaxf(mx, __shfl_xor(mx, msk, 32));
      const float mnew = fmaxf(mrow[i], mx);
      scale[i] = __expf(mrow[i] - mnew);
      p0[i] = __expf(s0[i] - mnew);
      p1[i] = __expf(s1[i] - mnew);
      float sum = p0[i] + p1[i];
#pragma unroll
      for (int msk = 1; msk < 16; msk <<= 1) sum += __shfl_xor(sum, msk, 32);
      lrow[i] = lrow[i] * scale[i] + sum;
      mrow[i] = mnew;
    }
#pragma unroll
    for (int t = 0; t < 8; ++t)
#pragma unroll
      for (int i = 0; i < 8; ++i) oacc[t][i] *= scale[i];
    // Transpose P (C-layout -> A-layout) through per-wave LDS tile.
#pragma unroll
    for (int i = 0; i < 8; ++i) {
      const int m = half * 8 + i;
      Pt[wave][m * 40 + n]      = f2bf(p0[i]);
      Pt[wave][m * 40 + 16 + n] = f2bf(p1[i]);
    }
    __builtin_amdgcn_wave_barrier();
    asm volatile("s_wait_dscnt 0x0" ::: "memory");  // wave-private tile; DS is in-order per wave
    const v16bf pf = load_a_frag(&Pt[wave][0], 40, lane);
    // O[16x128] += P[16x32] * V_chunk[32x128]
#pragma unroll
    for (int t = 0; t < 8; ++t) {
      v16bf bvv = load_bt_frag(&VtT[(t * 16) * 40], 40, lane);
      oacc[t] = __builtin_amdgcn_wmma_f32_16x16x32_bf16(false, pf, false, bvv, (short)0, oacc[t], false, false);
    }
  }
  // Normalize and store into [B,S,D] layout (bf16) for the output projection.
#pragma unroll
  for (int t = 0; t < 8; ++t) {
#pragma unroll
    for (int i = 0; i < 8; ++i) {
      const int m = half * 8 + i;
      const int s = q0 + m;
      const float v = oacc[t][i] / lrow[i];
      attnb[((size_t)(b * 2048 + s)) * 2048 + h * 128 + t * 16 + n] = f2bf(v);
    }
  }
}

extern "C" void kernel_launch(void* const* d_in, const int* in_sizes, int n_in,
                              void* d_out, int out_size, void* d_ws, size_t ws_size,
                              hipStream_t stream) {
  const float* x      = (const float*)d_in[0];
  const float* past_k = (const float*)d_in[1];
  const float* past_v = (const float*)d_in[2];
  const float* qkv_w  = (const float*)d_in[3];
  const float* qkv_b  = (const float*)d_in[4];
  const float* out_w  = (const float*)d_in[5];
  const float* out_b  = (const float*)d_in[6];
  float* outp = (float*)d_out;                 // [2,2048,2048]
  float* kout = outp + 8388608;                // [2,16,4096,128]
  float* vout = kout + 16777216;               // [2,16,4096,128]
  char* ws = (char*)d_ws;
  __bf16* xb    = (__bf16*)(ws + 0);
  __bf16* wqkv  = (__bf16*)(ws + 16777216);
  __bf16* wout  = (__bf16*)(ws + 41943040);
  __bf16* qb    = (__bf16*)(ws + 50331648);
  __bf16* kb    = (__bf16*)(ws + 67108864);
  __bf16* vb    = (__bf16*)(ws + 100663296);
  __bf16* attnb = (__bf16*)(ws + 134217728);

  f32_to_bf16_kernel<<<2048, 256, 0, stream>>>(x, xb, 2097152);
  f32_to_bf16_kernel<<<2048, 256, 0, stream>>>(qkv_w, wqkv, 3145728);
  f32_to_bf16_kernel<<<2048, 256, 0, stream>>>(out_w, wout, 1048576);
  copy_past_kernel<<<2048, 256, 0, stream>>>(past_k, kout, kb, 2097152);
  copy_past_kernel<<<2048, 256, 0, stream>>>(past_v, vout, vb, 2097152);
  gemm_bf16_kernel<0><<<dim3(48, 32), 256, 0, stream>>>(
      xb, wqkv, qkv_b, nullptr, kout, vout, qb, kb, vb, 2048);
  attn_kernel<<<dim3(16, 16, 2), 256, 0, stream>>>(qb, kb, vb, attnb);
  gemm_bf16_kernel<1><<<dim3(16, 32), 256, 0, stream>>>(
      attnb, wout, out_b, outp, nullptr, nullptr, nullptr, nullptr, nullptr, 2048);
}
